// SequenceToPairBlock_3539053052296
// MI455X (gfx1250) — compile-verified, matrix operand order checked by reading, not run on hardware
//
#include <hip/hip_runtime.h>
#include <hip/hip_bf16.h>
#include <math.h>

typedef __attribute__((ext_vector_type(16))) _Float16 v16h;
typedef __attribute__((ext_vector_type(8)))  float    v8f;

#define NUM_HEADS 32
#define HEAD_DIM  128
#define POOL      16
#define SP        512
#define D_MODEL   1536
#define HD        4096   /* NUM_HEADS*HEAD_DIM */
#define NPOS      1024   /* 2*SP */

/* ---- workspace layout (bytes) ---- */
#define OFF_XNORM 0ull
#define OFF_XGELU (OFF_XNORM + 512ull*1536*4)
#define OFF_Q     (OFF_XGELU + 512ull*1536*4)
#define OFF_K     (OFF_Q     + 512ull*4096*4)
#define OFF_POS   (OFF_K     + 512ull*4096*4)
#define OFF_YQ    (OFF_POS   + 1024ull*4096*4)
#define OFF_YK    (OFF_YQ    + 512ull*128*4)
#define OFF_TQ    (OFF_YK    + 512ull*128*4)          /* f16: 32*512*1024 */
#define OFF_TK    (OFF_TQ    + 32ull*512*1024*2)
#define OFF_A     (OFF_TK    + 32ull*512*1024*2)      /* f32: 512*512*32  */

/* ================= WMMA helpers (16x16x32 f16 -> f32) ================= */

__device__ __forceinline__ v8f wmma_f16(v16h a, v16h b, v8f c) {
  // (neg_a, A, neg_b, B, c_mod, C, reuse_a, reuse_b)
  return __builtin_amdgcn_wmma_f32_16x16x32_f16(false, a, false, b, (short)0, c,
                                                false, false);
}

/* Fragment load from 32 k-contiguous halves in LDS.
 * Caller passes p = base_of_32_halves + ((lane&16)>>1)  (hi half-wave -> +8).
 * Lane then reads runs [0..7] and [16..23]  => 2x ds_load_b128. */
__device__ __forceinline__ v16h load_frag(const _Float16* p) {
  v16h f;
#pragma unroll
  for (int j = 0; j < 8; ++j) {
    f[j]     = p[j];
    f[8 + j] = p[16 + j];
  }
  return f;
}

/* ================= 1. pool + RMSNorm + GELU ================= */

__global__ __launch_bounds__(256) void pool_rms_gelu_kernel(
    const float* __restrict__ x, const float* __restrict__ w_rms,
    float* __restrict__ xnorm, float* __restrict__ xgelu) {
  const int sp  = blockIdx.x;
  const int tid = threadIdx.x;
  float pooled[6];
  float ss = 0.f;
#pragma unroll
  for (int j = 0; j < 6; ++j) {
    const int col = tid + j * 256;
    const float* base = x + (size_t)sp * POOL * D_MODEL + col;
    float s = 0.f;
#pragma unroll
    for (int r = 0; r < POOL; ++r) s += base[(size_t)r * D_MODEL];
    s *= (1.0f / POOL);
    pooled[j] = s;
    ss += s * s;
  }
  __shared__ float red[256];
  red[tid] = ss;
  __syncthreads();
  for (int off = 128; off > 0; off >>= 1) {
    if (tid < off) red[tid] += red[tid + off];
    __syncthreads();
  }
  const float rstd = rsqrtf(red[0] * (1.0f / D_MODEL) + 1e-6f);
#pragma unroll
  for (int j = 0; j < 6; ++j) {
    const int col = tid + j * 256;
    const float xn = pooled[j] * rstd * w_rms[col];
    xnorm[(size_t)sp * D_MODEL + col] = xn;
    xgelu[(size_t)sp * D_MODEL + col] =
        0.5f * xn * (1.0f + erff(xn * 0.70710678118654752f));
  }
}

/* ================= 2. positional encoding ================= */

__global__ __launch_bounds__(256) void pos_enc_kernel(
    const float* __restrict__ W_pos, const float* __restrict__ b_pos,
    float* __restrict__ pos_enc) {
  const int p   = blockIdx.x;
  const int tid = threadIdx.x;
  const float dist = fabsf((float)(p - SP));
  const float sgn  = (p > SP) ? 1.0f : ((p < SP) ? -1.0f : 0.0f);
  const float log_step = logf((float)(SP - NUM_HEADS + 1)) / (float)NUM_HEADS;
  float acc[16];
#pragma unroll
  for (int j = 0; j < 16; ++j) acc[j] = 0.f;
  for (int f = 0; f < NUM_HEADS; ++f) {
    const float cw = (float)f + expf((float)f * log_step);
    if (cw > dist) {
      const float* w0 = W_pos + (size_t)f * HD;
      const float* w1 = W_pos + (size_t)(NUM_HEADS + f) * HD;
#pragma unroll
      for (int j = 0; j < 16; ++j) {
        const int col = tid + j * 256;
        acc[j] += w0[col] + sgn * w1[col];
      }
    }
  }
#pragma unroll
  for (int j = 0; j < 16; ++j) {
    const int col = tid + j * 256;
    pos_enc[(size_t)p * HD + col] = acc[j] + b_pos[col];
  }
}

/* ============ tiling parameters ============ */
#define GBM 128
#define GBN 64
#define GBK 64
#define LDKA 72 /* GBK+8 halves: per-row (A) / per-column (B) k-stride */
#define LDKP 40 /* 32+8, pair kernel */

/* ================= 3. generic f32 GEMM via f16 WMMA ================= */
/* C[M,N] = A[M,K] @ B[K,N], row-major. sA[m][k], sB column-major [n][k]. */

__global__ __launch_bounds__(256) void gemm_wmma_kernel(
    const float* __restrict__ A, const float* __restrict__ B,
    float* __restrict__ C, int M, int N, int K) {
  __shared__ _Float16 sA[GBM * LDKA];
  __shared__ _Float16 sB[GBN * LDKA];
  const int tid = threadIdx.x, lane = tid & 31, wave = tid >> 5;
  const int m0 = blockIdx.y * GBM;
  const int n0 = blockIdx.x * GBN;
  const int khalf = (lane & 16) >> 1;
  v8f acc[4] = {};

  for (int k0 = 0; k0 < K; k0 += GBK) {
#pragma unroll
    for (int i = 0; i < 8; ++i) { /* A tile: 128 x 64, k-contiguous */
      const int vec = tid + i * 256;
      const int row = vec >> 4;
      const int cv  = (vec & 15) << 2;
      const float4 av = *(const float4*)(A + (size_t)(m0 + row) * K + k0 + cv);
      _Float16* dst = sA + row * LDKA + cv;
      dst[0] = (_Float16)av.x; dst[1] = (_Float16)av.y;
      dst[2] = (_Float16)av.z; dst[3] = (_Float16)av.w;
    }
#pragma unroll
    for (int i = 0; i < 4; ++i) { /* B tile: 64k x 64n -> column-major LDS */
      const int g  = tid + i * 256;  /* 0..1023 */
      const int n  = g & 63;
      const int k4 = (g >> 6) << 2;
      const float* bp = B + (size_t)(k0 + k4) * N + n0 + n;
      _Float16* dst = sB + n * LDKA + k4;
      dst[0] = (_Float16)bp[0];
      dst[1] = (_Float16)bp[(size_t)N];
      dst[2] = (_Float16)bp[2 * (size_t)N];
      dst[3] = (_Float16)bp[3 * (size_t)N];
    }
    __syncthreads();
    const _Float16* arow = sA + (wave * 16 + (lane & 15)) * LDKA + khalf;
#pragma unroll
    for (int kk = 0; kk < 2; ++kk) {
      const v16h afrag = load_frag(arow + kk * 32);
#pragma unroll
      for (int t = 0; t < 4; ++t) {
        const _Float16* bcol =
            sB + (t * 16 + (lane & 15)) * LDKA + khalf + kk * 32;
        acc[t] = wmma_f16(afrag, load_frag(bcol), acc[t]);
      }
    }
    __syncthreads();
  }
  const int mrow  = m0 + wave * 16 + 8 * (lane >> 4);
  const int ncol0 = n0 + (lane & 15);
#pragma unroll
  for (int t = 0; t < 4; ++t)
#pragma unroll
    for (int v = 0; v < 8; ++v)
      C[(size_t)(mrow + v) * N + ncol0 + t * 16] = acc[t][v];
}

/* ====== 4. term GEMM: term[h] = (Q_h + bias_h) @ pos_enc_h^T, f16 out ====== */

__global__ __launch_bounds__(256) void term_wmma_kernel(
    const float* __restrict__ QK, const float* __restrict__ rbias,
    const float* __restrict__ pos_enc, _Float16* __restrict__ term) {
  __shared__ _Float16 sA[GBM * LDKA];
  __shared__ _Float16 sB[GBN * LDKA];
  const int tid = threadIdx.x, lane = tid & 31, wave = tid >> 5;
  const int h  = blockIdx.z;
  const int m0 = blockIdx.y * GBM;
  const int n0 = blockIdx.x * GBN;
  const int khalf = (lane & 16) >> 1;
  v8f acc[4] = {};

  for (int k0 = 0; k0 < HEAD_DIM; k0 += GBK) {
#pragma unroll
    for (int i = 0; i < 8; ++i) { /* A: QK[m, h*128+kc] + rbias */
      const int vec = tid + i * 256;
      const int row = vec >> 4;
      const int cv  = (vec & 15) << 2;
      const float4 av =
          *(const float4*)(QK + (size_t)(m0 + row) * HD + h * HEAD_DIM + k0 + cv);
      const float4 bb = *(const float4*)(rbias + h * HEAD_DIM + k0 + cv);
      _Float16* dst = sA + row * LDKA + cv;
      dst[0] = (_Float16)(av.x + bb.x); dst[1] = (_Float16)(av.y + bb.y);
      dst[2] = (_Float16)(av.z + bb.z); dst[3] = (_Float16)(av.w + bb.w);
    }
#pragma unroll
    for (int i = 0; i < 4; ++i) { /* B col-major: sB[p][kc] = pos_enc[p, h*128+kc] */
      const int vec = tid + i * 256;
      const int pl  = vec >> 4;
      const int kcv = (vec & 15) << 2;
      const float4 bv = *(const float4*)(pos_enc + (size_t)(n0 + pl) * HD +
                                         h * HEAD_DIM + k0 + kcv);
      _Float16* dst = sB + pl * LDKA + kcv;
      dst[0] = (_Float16)bv.x; dst[1] = (_Float16)bv.y;
      dst[2] = (_Float16)bv.z; dst[3] = (_Float16)bv.w;
    }
    __syncthreads();
    const _Float16* arow = sA + (wave * 16 + (lane & 15)) * LDKA + khalf;
#pragma unroll
    for (int kk = 0; kk < 2; ++kk) {
      const v16h afrag = load_frag(arow + kk * 32);
#pragma unroll
      for (int t = 0; t < 4; ++t) {
        const _Float16* bcol =
            sB + (t * 16 + (lane & 15)) * LDKA + khalf + kk * 32;
        acc[t] = wmma_f16(afrag, load_frag(bcol), acc[t]);
      }
    }
    __syncthreads();
  }
  const int mrow = m0 + wave * 16 + 8 * (lane >> 4);
  const int pc0  = n0 + (lane & 15);
  _Float16* tbase = term + (size_t)h * SP * NPOS;
#pragma unroll
  for (int t = 0; t < 4; ++t)
#pragma unroll
    for (int v = 0; v < 8; ++v)
      tbase[(size_t)(mrow + v) * NPOS + pc0 + t * 16] = (_Float16)acc[t][v];
}

/* ===== 5. a[q,k,h] = Q_h·K_h + 0.5*(termq[h][q][k-q+512] + termk[h][k][q-k+512]) */

__global__ __launch_bounds__(256) void attn_a_kernel(
    const float* __restrict__ Q, const float* __restrict__ Kmat,
    const _Float16* __restrict__ termq, const _Float16* __restrict__ termk,
    float* __restrict__ Aout) {
  __shared__ _Float16 sA[GBM * LDKA];
  __shared__ _Float16 sB[GBN * LDKA];
  const int tid = threadIdx.x, lane = tid & 31, wave = tid >> 5;
  const int h  = blockIdx.z;
  const int m0 = blockIdx.y * GBM;
  const int n0 = blockIdx.x * GBN;
  const int khalf = (lane & 16) >> 1;
  v8f acc[4] = {};

  for (int k0 = 0; k0 < HEAD_DIM; k0 += GBK) {
#pragma unroll
    for (int i = 0; i < 8; ++i) { /* A: Q[m, h*128+kc] */
      const int vec = tid + i * 256;
      const int row = vec >> 4;
      const int cv  = (vec & 15) << 2;
      const float4 av =
          *(const float4*)(Q + (size_t)(m0 + row) * HD + h * HEAD_DIM + k0 + cv);
      _Float16* dst = sA + row * LDKA + cv;
      dst[0] = (_Float16)av.x; dst[1] = (_Float16)av.y;
      dst[2] = (_Float16)av.z; dst[3] = (_Float16)av.w;
    }
#pragma unroll
    for (int i = 0; i < 4; ++i) { /* B col-major: sB[n][kc] = Kmat[n, h*128+kc] */
      const int vec = tid + i * 256;
      const int nl  = vec >> 4;
      const int kcv = (vec & 15) << 2;
      const float4 bv = *(const float4*)(Kmat + (size_t)(n0 + nl) * HD +
                                         h * HEAD_DIM + k0 + kcv);
      _Float16* dst = sB + nl * LDKA + kcv;
      dst[0] = (_Float16)bv.x; dst[1] = (_Float16)bv.y;
      dst[2] = (_Float16)bv.z; dst[3] = (_Float16)bv.w;
    }
    __syncthreads();
    const _Float16* arow = sA + (wave * 16 + (lane & 15)) * LDKA + khalf;
#pragma unroll
    for (int kk = 0; kk < 2; ++kk) {
      const v16h afrag = load_frag(arow + kk * 32);
#pragma unroll
      for (int t = 0; t < 4; ++t) {
        const _Float16* bcol =
            sB + (t * 16 + (lane & 15)) * LDKA + khalf + kk * 32;
        acc[t] = wmma_f16(afrag, load_frag(bcol), acc[t]);
      }
    }
    __syncthreads();
  }
  const int mbase = m0 + wave * 16 + 8 * (lane >> 4);
  const _Float16* tq = termq + (size_t)h * SP * NPOS;
  const _Float16* tk = termk + (size_t)h * SP * NPOS;
#pragma unroll
  for (int t = 0; t < 4; ++t) {
#pragma unroll
    for (int v = 0; v < 8; ++v) {
      const int m = mbase + v;
      const int n = n0 + t * 16 + (lane & 15);
      const int p1 = n - m + SP; /* shift identity: rel[q,k] = term[q, k-q+512] */
      const int p2 = m - n + SP;
      const float val = acc[t][v] +
          0.5f * ((float)tq[(size_t)m * NPOS + p1] +
                  (float)tk[(size_t)n * NPOS + p2]);
      Aout[((size_t)m * SP + n) * NUM_HEADS + h] = val;
    }
  }
}

/* ===== 6. pair = a @ W_pair + b_pair + y_q[q] + y_k[k] ; K=32 exact ===== */

#define PM 128

__global__ __launch_bounds__(256) void pair_final_kernel(
    const float* __restrict__ Ain, const float* __restrict__ W_pair,
    const float* __restrict__ b_pair, const float* __restrict__ yq,
    const float* __restrict__ yk, float* __restrict__ out) {
  __shared__ _Float16 sA[PM * LDKP];
  __shared__ _Float16 sB[128 * LDKP];
  __shared__ float sbias[128];
  const int tid = threadIdx.x, lane = tid & 31, wave = tid >> 5;
  const int khalf = (lane & 16) >> 1;
  const size_t r0 = (size_t)blockIdx.x * PM;

#pragma unroll
  for (int i = 0; i < 4; ++i) { /* A tile: 128 rows x 32 (K), k-contiguous */
    const int vec = tid + i * 256;
    const int row = vec >> 3;
    const int cv  = (vec & 7) << 2;
    const float4 av = *(const float4*)(Ain + (r0 + row) * 32 + cv);
    _Float16* dst = sA + row * LDKP + cv;
    dst[0] = (_Float16)av.x; dst[1] = (_Float16)av.y;
    dst[2] = (_Float16)av.z; dst[3] = (_Float16)av.w;
  }
#pragma unroll
  for (int i = 0; i < 4; ++i) { /* B: W_pair 32x128 -> column-major LDS */
    const int g  = tid + i * 256;  /* 0..1023 */
    const int n  = g & 127;
    const int k4 = (g >> 7) << 2;
    const float* bp = W_pair + (size_t)k4 * 128 + n;
    _Float16* dst = sB + n * LDKP + k4;
    dst[0] = (_Float16)bp[0];
    dst[1] = (_Float16)bp[128];
    dst[2] = (_Float16)bp[256];
    dst[3] = (_Float16)bp[384];
  }
  if (tid < 128) sbias[tid] = b_pair[tid];
  __syncthreads();

  const v16h afrag =
      load_frag(sA + (wave * 16 + (lane & 15)) * LDKP + khalf);
  const v8f zero = {};
  v8f acc[8];
#pragma unroll
  for (int t = 0; t < 8; ++t) {
    const _Float16* bcol = sB + (t * 16 + (lane & 15)) * LDKP + khalf;
    acc[t] = wmma_f16(afrag, load_frag(bcol), zero);
  }
  const int mloc = wave * 16 + 8 * (lane >> 4);
#pragma unroll
  for (int t = 0; t < 8; ++t) {
#pragma unroll
    for (int v = 0; v < 8; ++v) {
      const size_t r = r0 + mloc + v;
      const int q = (int)(r >> 9);
      const int k = (int)(r & 511);
      const int col = t * 16 + (lane & 15);
      out[r * 128 + col] = acc[t][v] + sbias[col] +
                           yq[(size_t)q * 128 + col] + yk[(size_t)k * 128 + col];
    }
  }
}

/* ================= launch ================= */

extern "C" void kernel_launch(void* const* d_in, const int* in_sizes, int n_in,
                              void* d_out, int out_size, void* d_ws,
                              size_t ws_size, hipStream_t stream) {
  (void)in_sizes; (void)n_in; (void)out_size; (void)ws_size;
  const float* x        = (const float*)d_in[0];
  const float* w_rms    = (const float*)d_in[1];
  const float* W_q      = (const float*)d_in[2];
  const float* W_k      = (const float*)d_in[3];
  const float* W_pos    = (const float*)d_in[4];
  const float* b_pos    = (const float*)d_in[5];
  const float* q_r_bias = (const float*)d_in[6];
  const float* k_r_bias = (const float*)d_in[7];
  const float* W_yq     = (const float*)d_in[8];
  const float* W_yk     = (const float*)d_in[9];
  const float* W_pair   = (const float*)d_in[10];
  const float* b_pair   = (const float*)d_in[11];
  float* out = (float*)d_out;

  char* ws = (char*)d_ws;
  float*    xnorm = (float*)(ws + OFF_XNORM);
  float*    xgelu = (float*)(ws + OFF_XGELU);
  float*    q_ws  = (float*)(ws + OFF_Q);
  float*    k_ws  = (float*)(ws + OFF_K);
  float*    pos   = (float*)(ws + OFF_POS);
  float*    yq_ws = (float*)(ws + OFF_YQ);
  float*    yk_ws = (float*)(ws + OFF_YK);
  _Float16* tq_ws = (_Float16*)(ws + OFF_TQ);
  _Float16* tk_ws = (_Float16*)(ws + OFF_TK);
  float*    a_ws  = (float*)(ws + OFF_A);

  pool_rms_gelu_kernel<<<SP, 256, 0, stream>>>(x, w_rms, xnorm, xgelu);
  pos_enc_kernel<<<NPOS, 256, 0, stream>>>(W_pos, b_pos, pos);

  gemm_wmma_kernel<<<dim3(HD / GBN, SP / GBM), 256, 0, stream>>>(
      xnorm, W_q, q_ws, SP, HD, D_MODEL);
  gemm_wmma_kernel<<<dim3(HD / GBN, SP / GBM), 256, 0, stream>>>(
      xnorm, W_k, k_ws, SP, HD, D_MODEL);
  gemm_wmma_kernel<<<dim3(HEAD_DIM / GBN, SP / GBM), 256, 0, stream>>>(
      xgelu, W_yq, yq_ws, SP, HEAD_DIM, D_MODEL);
  gemm_wmma_kernel<<<dim3(HEAD_DIM / GBN, SP / GBM), 256, 0, stream>>>(
      xgelu, W_yk, yk_ws, SP, HEAD_DIM, D_MODEL);

  term_wmma_kernel<<<dim3(NPOS / GBN, SP / GBM, NUM_HEADS), 256, 0, stream>>>(
      q_ws, q_r_bias, pos, tq_ws);
  term_wmma_kernel<<<dim3(NPOS / GBN, SP / GBM, NUM_HEADS), 256, 0, stream>>>(
      k_ws, k_r_bias, pos, tk_ws);

  attn_a_kernel<<<dim3(SP / GBN, SP / GBM, NUM_HEADS), 256, 0, stream>>>(
      q_ws, k_ws, tq_ws, tk_ws, a_ws);

  pair_final_kernel<<<(SP * SP) / PM, 256, 0, stream>>>(
      a_ws, W_pair, b_pair, yq_ws, yk_ws, out);
}
